// TensorProduct_40338332844517
// MI455X (gfx1250) — matrix operand order repaired and checked by reading, not compiled, not run on hardware
//
#include <hip/hip_runtime.h>
#include <hip/hip_bf16.h>

// ---------------------------------------------------------------------------
// e3nn tensor product (64x0e+64x1o) x (1x0e+1x1o) -> 64x0e+64x1o+64x1e
// Reduced to 9 [16x64]x[64x64] GEMM groups per 16-element batch tile,
// executed with v_wmma_f32_16x16x32_f16 on gfx1250 (wave32).
// LDS rows padded to 72 f16 (144 B): 16B-aligned rows -> ds_load_b128
// fragment gathers, and 36-dword stride makes b128 gathers bank-conflict-free.
// ---------------------------------------------------------------------------

typedef __attribute__((ext_vector_type(16))) _Float16 v16h;
typedef __attribute__((ext_vector_type(8)))  float    v8f;

#define S_MAIN 0.08838834764831845f   /* 1/sqrt(128): 0e-pathA, 1o, 1e */
#define S_0EB  0.05103103630798287f   /* 1/sqrt(384): 0e-pathB (w110)  */

#define WROW 72   /* padded LDS row stride in f16 units (144 B, 16B aligned) */

union F16Frag { v16h v; uint4 q[2]; };

__device__ __forceinline__ v8f wmma16(v16h a, v16h b, v8f c) {
  return __builtin_amdgcn_wmma_f32_16x16x32_f16(
      /*neg_a=*/false, a, /*neg_b=*/false, b,
      /*c_mod=*/(short)0, c, /*reuse_a=*/false, /*reuse_b=*/false);
}

// A-matrix 16x32 f16 layout (ISA 7.12.2): lanes 0-15 hold K 0..7 (VGPR0-3)
// and K 16..23 (VGPR4-7); lanes 16-31 hold K 8..15 and 24..31. Pairs per VGPR.
// Dword base = 4*hi + 16*ks; groups at +0..3 and +8..11  ->  q-index hi+4ks, +2.
__device__ __forceinline__ v16h load_frag_A(const uint4* row4, int hi, int ks) {
  F16Frag r;
  const int bq = hi + 4 * ks;
  r.q[0] = row4[bq];
  r.q[1] = row4[bq + 2];
  return r.v;
}

// B-matrix 32x16 f16 layout (same K order as A):
// lanes 0-15 hold K 0..7 and K 16..23, lanes 16-31 hold K 8..15 and 24..31.
// Dword base = 4*hi + 16*ks; groups at +0..3 and +8..11 -> q-index hi+4ks, +2.
__device__ __forceinline__ v16h load_frag_B(const uint4* row4, int hi, int ks) {
  F16Frag r;
  const int bq = hi + 4 * ks;
  r.q[0] = row4[bq];
  r.q[1] = row4[bq + 2];
  return r.v;
}

__global__ __launch_bounds__(128)
void tp_wmma(const float* __restrict__ x1, const float* __restrict__ x2,
             const float* __restrict__ w, float* __restrict__ out) {
  extern __shared__ char smem[];
  _Float16* ldsw = (_Float16*)smem;            // 5*64*WROW f16 (W^T, n-major)
  _Float16* feat = ldsw + 5 * 64 * WROW;       // 8*64*WROW f16 (features)
  float*    x2s  = (float*)(feat + 8 * 64 * WROW); // 64*4 f32

  const int t  = threadIdx.x;
  const int zb = blockIdx.x * 64;

  // ---- stage weights: ldsw[p][n][u] = (f16) w[p][u][0][n]
  for (int i = t; i < 5 * 64 * 64; i += 128) {
    int p  = i >> 12;
    int u  = (i >> 6) & 63;
    int nn = i & 63;
    ldsw[(p * 64 + nn) * WROW + u] = (_Float16)w[i];
  }
  // ---- stage x2 tile
  if (t < 64) {
    const float* src = x2 + (size_t)(zb + t) * 4;
    float* dst = x2s + t * 4;
    dst[0] = src[0]; dst[1] = src[1]; dst[2] = src[2]; dst[3] = src[3];
  }
  __syncthreads();

  // ---- compute the 8 feature vectors per element, store f16 in LDS
  for (int i = t; i < 64 * 64; i += 128) {
    int z = i >> 6, u = i & 63;
    const float* r = x1 + (size_t)(zb + z) * 256;
    float x10 = r[u];
    float a0 = r[64 + 3 * u + 0];
    float a1 = r[64 + 3 * u + 1];
    float a2 = r[64 + 3 * u + 2];
    const float* q = x2s + z * 4;
    float b0 = q[1], b1 = q[2], b2 = q[3];
    float dt = a0 * b0 + a1 * b1 + a2 * b2;       // w110 dot
    float c0 = a1 * b2 - a2 * b1;                 // w111 cross
    float c1 = a2 * b0 - a0 * b2;
    float c2 = a0 * b1 - a1 * b0;
    int zr = z * WROW + u;
    feat[0 * 64 * WROW + zr] = (_Float16)x10;
    feat[1 * 64 * WROW + zr] = (_Float16)dt;
    feat[2 * 64 * WROW + zr] = (_Float16)a0;
    feat[3 * 64 * WROW + zr] = (_Float16)a1;
    feat[4 * 64 * WROW + zr] = (_Float16)a2;
    feat[5 * 64 * WROW + zr] = (_Float16)c0;
    feat[6 * 64 * WROW + zr] = (_Float16)c1;
    feat[7 * 64 * WROW + zr] = (_Float16)c2;
  }
  __syncthreads();

  // ---- WMMA phase: each wave owns 16 batch elements
  const int lane = t & 31;
  const int wave = t >> 5;
  const int col  = lane & 15;
  const int hi   = lane >> 4;
  const int wz   = wave * 16;
  const int zrow = wz + col;      // A-fragment row for this lane

  const _Float16* featB = feat;
  const _Float16* ldswB = ldsw;
#define FEAT_ROW(f) ((const uint4*)(featB + ((f) * 64 + zrow) * WROW))
#define W_ROW(p, nn) ((const uint4*)(ldswB + ((p) * 64 + (nn)) * WROW))

  // ======== group 1e: R4k = f(5+k) x W4 ========
  {
    v16h a5[2], a6[2], a7[2];
#pragma unroll
    for (int ks = 0; ks < 2; ++ks) {
      a5[ks] = load_frag_A(FEAT_ROW(5), hi, ks);
      a6[ks] = load_frag_A(FEAT_ROW(6), hi, ks);
      a7[ks] = load_frag_A(FEAT_ROW(7), hi, ks);
    }
    for (int nt = 0; nt < 4; ++nt) {
      int nn = nt * 16 + col;
      v8f e0 = {}, e1 = {}, e2 = {};
#pragma unroll
      for (int ks = 0; ks < 2; ++ks) {
        v16h b = load_frag_B(W_ROW(4, nn), hi, ks);
        e0 = wmma16(a5[ks], b, e0);
        e1 = wmma16(a6[ks], b, e1);
        e2 = wmma16(a7[ks], b, e2);
      }
#pragma unroll
      for (int g = 0; g < 8; ++g) {
        int z = wz + 8 * hi + g;                 // C layout: M = g (+8 if hi)
        size_t ob = (size_t)(zb + z) * 448 + 256 + 3 * nn;
        out[ob + 0] = S_MAIN * e0[g];
        out[ob + 1] = S_MAIN * e1[g];
        out[ob + 2] = S_MAIN * e2[g];
      }
    }
  }

  // f0 fragments shared by 0e and 1o groups
  v16h a0[2];
#pragma unroll
  for (int ks = 0; ks < 2; ++ks) a0[ks] = load_frag_A(FEAT_ROW(0), hi, ks);

  // ======== group 0e: R0 = f0 x W0, R2 = f1 x W3 ========
  {
    v16h a1f[2];
#pragma unroll
    for (int ks = 0; ks < 2; ++ks) a1f[ks] = load_frag_A(FEAT_ROW(1), hi, ks);
    for (int nt = 0; nt < 4; ++nt) {
      int nn = nt * 16 + col;
      v8f r0 = {}, r2 = {};
#pragma unroll
      for (int ks = 0; ks < 2; ++ks) {
        v16h bb0 = load_frag_B(W_ROW(0, nn), hi, ks);
        v16h bb3 = load_frag_B(W_ROW(3, nn), hi, ks);
        r0 = wmma16(a0[ks], bb0, r0);
        r2 = wmma16(a1f[ks], bb3, r2);
      }
#pragma unroll
      for (int g = 0; g < 8; ++g) {
        int z = wz + 8 * hi + g;
        float x20 = x2s[z * 4];
        out[(size_t)(zb + z) * 448 + nn] = S_MAIN * x20 * r0[g] + S_0EB * r2[g];
      }
    }
  }

  // ======== group 1o: R1 = f0 x W1, R3k = f(2+k) x W2 ========
  {
    v16h a2[2], a3[2], a4[2];
#pragma unroll
    for (int ks = 0; ks < 2; ++ks) {
      a2[ks] = load_frag_A(FEAT_ROW(2), hi, ks);
      a3[ks] = load_frag_A(FEAT_ROW(3), hi, ks);
      a4[ks] = load_frag_A(FEAT_ROW(4), hi, ks);
    }
    for (int nt = 0; nt < 4; ++nt) {
      int nn = nt * 16 + col;
      v8f r1 = {}, q0 = {}, q1 = {}, q2 = {};
#pragma unroll
      for (int ks = 0; ks < 2; ++ks) {
        v16h bw1 = load_frag_B(W_ROW(1, nn), hi, ks);
        v16h bw2 = load_frag_B(W_ROW(2, nn), hi, ks);
        r1 = wmma16(a0[ks], bw1, r1);
        q0 = wmma16(a2[ks], bw2, q0);
        q1 = wmma16(a3[ks], bw2, q1);
        q2 = wmma16(a4[ks], bw2, q2);
      }
#pragma unroll
      for (int g = 0; g < 8; ++g) {
        int z = wz + 8 * hi + g;
        const float* q = x2s + z * 4;
        size_t ob = (size_t)(zb + z) * 448 + 64 + 3 * nn;
        out[ob + 0] = S_MAIN * (r1[g] * q[1] + q[0] * q0[g]);
        out[ob + 1] = S_MAIN * (r1[g] * q[2] + q[0] * q1[g]);
        out[ob + 2] = S_MAIN * (r1[g] * q[3] + q[0] * q2[g]);
      }
    }
  }
#undef FEAT_ROW
#undef W_ROW
}

// Scalar fallback for any batch remainder (unused for N = 200000).
__global__ void tp_tail(const float* __restrict__ x1, const float* __restrict__ x2,
                        const float* __restrict__ w, float* __restrict__ out, int z0) {
  int z  = z0 + blockIdx.x;
  int wo = threadIdx.x;                         // 64 threads = output channel
  const float* r = x1 + (size_t)z * 256;
  const float* q = x2 + (size_t)z * 4;
  float x20 = q[0], b0 = q[1], b1 = q[2], b2 = q[3];
  float s0 = 0.f, s1v = 0.f, sB = 0.f;
  float sq0 = 0.f, sq1 = 0.f, sq2 = 0.f;
  float se0 = 0.f, se1 = 0.f, se2 = 0.f;
  for (int u = 0; u < 64; ++u) {
    float x10 = r[u];
    float a0 = r[64 + 3 * u + 0];
    float a1 = r[64 + 3 * u + 1];
    float a2 = r[64 + 3 * u + 2];
    float dt = a0 * b0 + a1 * b1 + a2 * b2;
    float c0 = a1 * b2 - a2 * b1;
    float c1 = a2 * b0 - a0 * b2;
    float c2 = a0 * b1 - a1 * b0;
    float w0 = w[0 * 4096 + u * 64 + wo];
    float w1 = w[1 * 4096 + u * 64 + wo];
    float w2 = w[2 * 4096 + u * 64 + wo];
    float w3 = w[3 * 4096 + u * 64 + wo];
    float w4 = w[4 * 4096 + u * 64 + wo];
    s0  += w0 * x10;
    s1v += w1 * x10;
    sB  += w3 * dt;
    sq0 += w2 * a0; sq1 += w2 * a1; sq2 += w2 * a2;
    se0 += w4 * c0; se1 += w4 * c1; se2 += w4 * c2;
  }
  size_t ob = (size_t)z * 448;
  out[ob + wo] = S_MAIN * x20 * s0 + S_0EB * sB;
  out[ob + 64 + 3 * wo + 0] = S_MAIN * (s1v * b0 + x20 * sq0);
  out[ob + 64 + 3 * wo + 1] = S_MAIN * (s1v * b1 + x20 * sq1);
  out[ob + 64 + 3 * wo + 2] = S_MAIN * (s1v * b2 + x20 * sq2);
  out[ob + 256 + 3 * wo + 0] = S_MAIN * se0;
  out[ob + 256 + 3 * wo + 1] = S_MAIN * se1;
  out[ob + 256 + 3 * wo + 2] = S_MAIN * se2;
}

extern "C" void kernel_launch(void* const* d_in, const int* in_sizes, int n_in,
                              void* d_out, int out_size, void* d_ws, size_t ws_size,
                              hipStream_t stream) {
  (void)n_in; (void)out_size; (void)d_ws; (void)ws_size;
  const float* x1 = (const float*)d_in[0];
  const float* x2 = (const float*)d_in[1];
  const float* w  = (const float*)d_in[2];
  float* out = (float*)d_out;

  int n = in_sizes[0] / 256;        // batch size (200000)
  int tiles = n / 64;               // 64 elements per workgroup
  // LDS: (5*64 + 8*64) rows * WROW f16 + 64*4 f32
  size_t smem = (size_t)(13 * 64 * WROW) * 2 + 64 * 4 * 4;   // 120832 B
  if (tiles > 0)
    tp_wmma<<<tiles, 128, smem, stream>>>(x1, x2, w, out);
  int rem = n - tiles * 64;
  if (rem > 0)
    tp_tail<<<rem, 64, 0, stream>>>(x1, x2, w, out, tiles * 64);
}
